// Qwen35DecoderLayer_12146167513344
// MI455X (gfx1250) — matrix operand order repaired and checked by reading, not compiled, not run on hardware
//
#include <hip/hip_runtime.h>
#include <hip/hip_bf16.h>

typedef __attribute__((ext_vector_type(16))) __bf16 v16bf;
typedef __attribute__((ext_vector_type(8)))  float  v8f;

namespace {
constexpr int BB = 2, S = 4096, H = 16, DK = 128, DV = 128;
constexpr int C = 64, NCHUNK = S / C;
constexpr int DVT = 32, VT = DV / DVT;          // Dv split across blocks
constexpr int LDK  = 136;                        // stride for [*][128] bf16 (272B, 16B aligned)
constexpr int LD64 = 72;                         // stride for [*][64]  bf16 (144B, 16B aligned)
constexpr int LDA  = 68;                         // stride for [64][64] f32  (272B)
constexpr int LDU  = 36;                         // stride for [*][32]  f32  (144B)
constexpr float QSCALE = 0.08838834764831845f;   // Dk^-0.5
}

__device__ __forceinline__ v8f wmma_bf16(v16bf a, v16bf b, v8f c) {
  return __builtin_amdgcn_wmma_f32_16x16x32_bf16(false, a, false, b, (short)0, c, false, false);
}

// A fragment (16x32, MxK): lanes 0-15 row M=row0+lane, elems[0..7]=K(8*hi..),
// elems[8..15]=K(16+8*hi..), hi = lane>>4.  Two aligned 16B LDS reads.
template<int LD>
__device__ __forceinline__ v16bf load_a_frag(const __bf16* base, int row0, int k0, int lane) {
  const int l = lane & 15, hi = (lane >> 4) & 1;
  const __bf16* r = base + (row0 + l) * LD + k0;
  v16bf a;
#pragma unroll
  for (int j = 0; j < 8; ++j) { a[j] = r[8 * hi + j]; a[8 + j] = r[16 + 8 * hi + j]; }
  return a;
}

// B fragment (32x16, KxN) for X @ Y^T: Y stored row-major [N][K]; lane holds
// column N=col0+(lane&15), elems[0..7]=K(8*hi..), elems[8..15]=K(16+8*hi..).
template<int LD>
__device__ __forceinline__ v16bf load_b_frag(const __bf16* base, int col0, int k0, int lane) {
  const int l = lane & 15, hi = (lane >> 4) & 1;
  const __bf16* r = base + (col0 + l) * LD + k0;
  v16bf b;
#pragma unroll
  for (int j = 0; j < 8; ++j) { b[j] = r[8 * hi + j]; b[8 + j] = r[16 + 8 * hi + j]; }
  return b;
}

__global__ __launch_bounds__(256, 1)
void gdn_chunk_scan(const float* __restrict__ qp, const float* __restrict__ kp,
                    const float* __restrict__ vp, const float* __restrict__ gp,
                    const float* __restrict__ bp, float* __restrict__ outp) {
  __shared__ alignas(16) __bf16 k_lds[C * LDK];     // normalized k
  __shared__ alignas(16) __bf16 kb_kcd[C * LDK];    // k*beta, later k_cumdecay
  __shared__ alignas(16) __bf16 qg_lds[C * LDK];    // q*scale*exp(gc)
  __shared__ alignas(16) __bf16 kT[DK * LD64];      // (k*beta*exp(gc))^T, later k_dec^T
  __shared__ alignas(16) __bf16 stT[DVT * LDK];     // bf16 state^T [32][128]
  __shared__ alignas(16) __bf16 vbvn[DVT * LD64];   // (v*beta)^T, later v_new^T
  __shared__ alignas(16) __bf16 tb_attn[C * LD64];  // bf16 T, later attn
  __shared__ alignas(16) float  Af[C * LDA];        // A^T (column i of A = row i of Af)
  __shared__ alignas(16) float  Tf[C * LDA];        // published rows of T
  __shared__ alignas(16) float  u_vn[C * LDU];      // u, later v_new (f32)
  __shared__ alignas(16) float  st[DK * LDU];       // recurrent state [128][32] f32
  __shared__ alignas(16) float gbuf[C], bbuf[C], gcs[C], egs[C], iegs[C];

  const int tid  = threadIdx.x;
  const int wv   = tid >> 5, lane = tid & 31;
  const int b    = blockIdx.x / (H * VT);
  const int hv   = blockIdx.x % (H * VT);
  const int h    = hv / VT;
  const int v0   = (hv % VT) * DVT;
  const int cidx = tid & 63;        // owned column of T
  const int rg   = tid >> 6;        // owned row group: rows 16*rg .. 16*rg+15

  for (int i = tid; i < DK * LDU; i += 256) st[i] = 0.f;
  __syncthreads();

  for (int n = 0; n < NCHUNK; ++n) {
    const int s0 = n * C;
    // ---- gates ----
    if (tid < C) {
      const size_t gi = (size_t)(b * S + s0 + tid) * H + h;
      gbuf[tid] = gp[gi];
      bbuf[tid] = bp[gi];
    }
    __syncthreads();
    if (tid == 0) {
      float run = 0.f;
      for (int i = 0; i < C; ++i) { run += gbuf[i]; gcs[i] = run; }
    }
    __syncthreads();
    if (tid < C) { egs[tid] = __expf(gcs[tid]); iegs[tid] = __expf(-gcs[tid]); }
    __syncthreads();

    // ---- load + l2norm + stage bf16 operands ----
    {
      const int row = tid >> 2, sub = tid & 3;      // 4 threads per token row
      const size_t rbase = ((size_t)(b * S + s0 + row) * H + h) * (size_t)DK + sub * 32;
      const float4* q4 = reinterpret_cast<const float4*>(qp + rbase);
      const float4* k4 = reinterpret_cast<const float4*>(kp + rbase);
      float4 qv[8], kv[8];
      float sq = 0.f, sk = 0.f;
#pragma unroll
      for (int j = 0; j < 8; ++j) {
        qv[j] = q4[j]; kv[j] = k4[j];
        sq += qv[j].x * qv[j].x + qv[j].y * qv[j].y + qv[j].z * qv[j].z + qv[j].w * qv[j].w;
        sk += kv[j].x * kv[j].x + kv[j].y * kv[j].y + kv[j].z * kv[j].z + kv[j].w * kv[j].w;
      }
      sq += __shfl_xor(sq, 1); sq += __shfl_xor(sq, 2);
      sk += __shfl_xor(sk, 1); sk += __shfl_xor(sk, 2);
      const float rq = rsqrtf(sq + 1e-6f), rk = rsqrtf(sk + 1e-6f);
      const float beta = bbuf[row], eg = egs[row];
      const float qs = rq * QSCALE * eg, kbg = beta * eg;
#pragma unroll
      for (int j = 0; j < 8; ++j) {
        const int d = sub * 32 + 4 * j;
        const float ks[4] = {kv[j].x * rk, kv[j].y * rk, kv[j].z * rk, kv[j].w * rk};
        const float qsv[4] = {qv[j].x * qs, qv[j].y * qs, qv[j].z * qs, qv[j].w * qs};
#pragma unroll
        for (int e = 0; e < 4; ++e) {
          k_lds[row * LDK + d + e]  = (__bf16)ks[e];
          kb_kcd[row * LDK + d + e] = (__bf16)(ks[e] * beta);
          kT[(d + e) * LD64 + row]  = (__bf16)(ks[e] * kbg);
          qg_lds[row * LDK + d + e] = (__bf16)qsv[e];
        }
      }
      // value tile (transposed, *beta)
      const float4* v4 = reinterpret_cast<const float4*>(
          vp + ((size_t)(b * S + s0 + row) * H + h) * (size_t)DV + v0 + sub * 8);
#pragma unroll
      for (int j = 0; j < 2; ++j) {
        const float4 vvv = v4[j];
        const int c0 = sub * 8 + 4 * j;
        vbvn[(c0 + 0) * LD64 + row] = (__bf16)(vvv.x * beta);
        vbvn[(c0 + 1) * LD64 + row] = (__bf16)(vvv.y * beta);
        vbvn[(c0 + 2) * LD64 + row] = (__bf16)(vvv.z * beta);
        vbvn[(c0 + 3) * LD64 + row] = (__bf16)(vvv.w * beta);
      }
      if (n + 1 < NCHUNK) {   // prefetch next chunk (global_prefetch_b8)
        const size_t nbase = ((size_t)(b * S + s0 + C + row) * H + h) * (size_t)DK + sub * 32;
        __builtin_prefetch(qp + nbase, 0, 3);
        __builtin_prefetch(kp + nbase, 0, 3);
      }
    }
    __syncthreads();

    // ---- A = -(kb @ k^T) * decay, strict lower; stored transposed in Af ----
#pragma unroll
    for (int t = wv; t < 16; t += 8) {
      const int tm = (t >> 2) * 16, tn = (t & 3) * 16;
      v8f acc = {};
#pragma unroll
      for (int kk = 0; kk < DK; kk += 32)
        acc = wmma_bf16(load_a_frag<LDK>(kb_kcd, tm, kk, lane),
                        load_b_frag<LDK>(k_lds, tn, kk, lane), acc);
      const int nn = tn + (lane & 15), hi = (lane >> 4) & 1;
      const float ieg_n = iegs[nn];
      const float4* eg4 = reinterpret_cast<const float4*>(&egs[tm + 8 * hi]);
      const float4 e0 = eg4[0], e1 = eg4[1];
      const float em[8] = {e0.x, e0.y, e0.z, e0.w, e1.x, e1.y, e1.z, e1.w};
#pragma unroll
      for (int r = 0; r < 8; ++r) {
        const int mm = tm + r + 8 * hi;
        const float val = -acc[r] * em[r] * ieg_n;
        Af[nn * LDA + mm] = (nn < mm) ? val : 0.f;   // transposed: Af[row i] = A column i
      }
    }
    __syncthreads();

    // ---- T = (I - A)^{-1}: register-resident forward substitution ----
    {
      float xacc[16];
#pragma unroll
      for (int j = 0; j < 16; ++j) xacc[j] = ((16 * rg + j) == cidx) ? 1.f : 0.f;
      for (int o = 0; o < 4; ++o) {
#pragma unroll
        for (int ii = 0; ii < 16; ++ii) {
          const int i = 16 * o + ii;
          if (rg == o) Tf[i * LDA + cidx] = xacc[ii];   // publish final row i
          __syncthreads();
          const float xi = Tf[i * LDA + cidx];
          const float4* af4 = reinterpret_cast<const float4*>(&Af[i * LDA + 16 * rg]);
          const float4 a0 = af4[0], a1 = af4[1], a2 = af4[2], a3 = af4[3];
          const float av[16] = {a0.x, a0.y, a0.z, a0.w, a1.x, a1.y, a1.z, a1.w,
                                a2.x, a2.y, a2.z, a2.w, a3.x, a3.y, a3.z, a3.w};
#pragma unroll
          for (int j = 0; j < 16; ++j) {
            const int r = 16 * rg + j;
            xacc[j] += ((r > i) ? av[j] : 0.f) * xi;
          }
        }
      }
      __syncthreads();
#pragma unroll
      for (int j = 0; j < 16; ++j)          // bf16 T straight from registers
        tb_attn[(16 * rg + j) * LD64 + cidx] = (__bf16)xacc[j];
    }
    __syncthreads();

    // ---- u = T @ (v*beta) tile  and  kcd = T @ (kb*exp(g)) ----
    {
      const int tm = (wv >> 1) * 16, tn = (wv & 1) * 16;
      v8f acc = {};
#pragma unroll
      for (int kk = 0; kk < C; kk += 32)
        acc = wmma_bf16(load_a_frag<LD64>(tb_attn, tm, kk, lane),
                        load_b_frag<LD64>(vbvn, tn, kk, lane), acc);
      const int nn = tn + (lane & 15), hi = (lane >> 4) & 1;
#pragma unroll
      for (int r = 0; r < 8; ++r) u_vn[(tm + r + 8 * hi) * LDU + nn] = acc[r];
    }
#pragma unroll
    for (int t = wv; t < 32; t += 8) {
      const int tm = (t >> 3) * 16, tn = (t & 7) * 16;
      v8f acc = {};
#pragma unroll
      for (int kk = 0; kk < C; kk += 32)
        acc = wmma_bf16(load_a_frag<LD64>(tb_attn, tm, kk, lane),
                        load_b_frag<LD64>(kT, tn, kk, lane), acc);
      const int nn = tn + (lane & 15), hi = (lane >> 4) & 1;
#pragma unroll
      for (int r = 0; r < 8; ++r) kb_kcd[(tm + r + 8 * hi) * LDK + nn] = (__bf16)acc[r];
    }
    __syncthreads();

    // ---- attn = tril((qg @ k^T) * exp(-gc_j));  k_dec^T ----
#pragma unroll
    for (int t = wv; t < 16; t += 8) {
      const int tm = (t >> 2) * 16, tn = (t & 3) * 16;
      v8f acc = {};
#pragma unroll
      for (int kk = 0; kk < DK; kk += 32)
        acc = wmma_bf16(load_a_frag<LDK>(qg_lds, tm, kk, lane),
                        load_b_frag<LDK>(k_lds, tn, kk, lane), acc);
      const int nn = tn + (lane & 15), hi = (lane >> 4) & 1;
      const float ieg_n = iegs[nn];
#pragma unroll
      for (int r = 0; r < 8; ++r) {
        const int mm = tm + r + 8 * hi;
        const float val = acc[r] * ieg_n;
        tb_attn[mm * LD64 + nn] = (__bf16)((nn <= mm) ? val : 0.f);
      }
    }
    const float egl = egs[C - 1];
    for (int idx = tid; idx < C * DK; idx += 256) {
      const int i = idx >> 7, d = idx & (DK - 1);
      kT[d * LD64 + i] = (__bf16)((float)k_lds[i * LDK + d] * (egl * iegs[i]));
    }
    __syncthreads();

    // ---- inter-chunk: v' = kcd@state, attn_inter = qg@state ----
    for (int idx = tid; idx < DK * DVT; idx += 256) {
      const int d = idx >> 5, v = idx & 31;
      stT[v * LDK + d] = (__bf16)st[d * LDU + v];
    }
    __syncthreads();
    v8f accO = {};
    {
      const int tm = (wv >> 1) * 16, tn = (wv & 1) * 16;
      v8f acc1 = {};
#pragma unroll
      for (int kk = 0; kk < DK; kk += 32) {
        v16bf bb = load_b_frag<LDK>(stT, tn, kk, lane);
        acc1 = wmma_bf16(load_a_frag<LDK>(kb_kcd, tm, kk, lane), bb, acc1);
        accO = wmma_bf16(load_a_frag<LDK>(qg_lds, tm, kk, lane), bb, accO);
      }
      const int nn = tn + (lane & 15), hi = (lane >> 4) & 1;
#pragma unroll
      for (int r = 0; r < 8; ++r) {
        const int mm = tm + r + 8 * hi;
        const float vnew = u_vn[mm * LDU + nn] - acc1[r];
        u_vn[mm * LDU + nn] = vnew;
        vbvn[nn * LD64 + mm] = (__bf16)vnew;
      }
    }
    __syncthreads();

    // ---- out = attn_inter + attn @ v_new ----
    {
      const int tm = (wv >> 1) * 16, tn = (wv & 1) * 16;
#pragma unroll
      for (int kk = 0; kk < C; kk += 32)
        accO = wmma_bf16(load_a_frag<LD64>(tb_attn, tm, kk, lane),
                         load_b_frag<LD64>(vbvn, tn, kk, lane), accO);
      const int nn = tn + (lane & 15), hi = (lane >> 4) & 1;
#pragma unroll
      for (int r = 0; r < 8; ++r) {
        const int mm = tm + r + 8 * hi;
        outp[((size_t)(b * S + s0 + mm) * H + h) * (size_t)DV + v0 + nn] = accO[r];
      }
    }
    // ---- state = state*exp(g_last) + k_dec^T @ v_new ----
#pragma unroll
    for (int t = wv; t < 16; t += 8) {
      const int tm = (t >> 1) * 16, tn = (t & 1) * 16;
      v8f acc = {};
#pragma unroll
      for (int kk = 0; kk < C; kk += 32)
        acc = wmma_bf16(load_a_frag<LD64>(kT, tm, kk, lane),
                        load_b_frag<LD64>(vbvn, tn, kk, lane), acc);
      const int nn = tn + (lane & 15), hi = (lane >> 4) & 1;
#pragma unroll
      for (int r = 0; r < 8; ++r) {
        const int mm = tm + r + 8 * hi;
        st[mm * LDU + nn] = st[mm * LDU + nn] * egl + acc[r];
      }
    }
    __syncthreads();
  }
}

extern "C" void kernel_launch(void* const* d_in, const int* in_sizes, int n_in,
                              void* d_out, int out_size, void* d_ws, size_t ws_size,
                              hipStream_t stream) {
  (void)in_sizes; (void)n_in; (void)out_size; (void)d_ws; (void)ws_size;
  const float* q    = (const float*)d_in[0];
  const float* k    = (const float*)d_in[1];
  const float* v    = (const float*)d_in[2];
  const float* g    = (const float*)d_in[3];
  const float* beta = (const float*)d_in[4];
  float* out = (float*)d_out;
  gdn_chunk_scan<<<dim3(BB * H * VT), dim3(256), 0, stream>>>(q, k, v, g, beta, out);
}
